// SpikeRNNHAR_79216376807589
// MI455X (gfx1250) — matrix operand order repaired and checked by reading, not compiled, not run on hardware
//
#include <hip/hip_runtime.h>

// ---------------------------------------------------------------------------
// Persistent spiking-RNN kernel for MI455X (gfx1250, wave32, WMMA).
// 8 workgroups x 384 threads (12 wave32s); each WG owns 16 batch rows for all
// 2048 timesteps. Each wave owns one 16x16 output tile per layer. Per layer-
// phase: 6 v_wmma_f32_16x16x32_f16 split into TWO interleaved independent
// accumulation chains (depth 3) to halve D->C dependency latency. All
// recurrent weights (f16 B fragments, 96 VGPRs) are register-resident;
// membrane potentials live in VGPRs; only spikes + staged x live in LDS.
// 2 workgroup barriers per timestep.
// ---------------------------------------------------------------------------

typedef __attribute__((ext_vector_type(16))) _Float16 v16h;
typedef __attribute__((ext_vector_type(8)))  float    v8f;
typedef __attribute__((ext_vector_type(4)))  float    v4f;
typedef __attribute__((ext_vector_type(4)))  unsigned v4u;

#define TT    2048
#define HH    192
#define BT    16            // batch rows per workgroup (one M tile)
#define TPB   384           // 12 wave32s: wave w owns N-tile w for both layers
#define KB    6             // K blocks of 32 (192 = 6*32)
#define HSROW 400           // padded spike-row stride in bytes (192 f16 + pad)

// ---- LDS layout (bytes) ----
#define OFF_HS0   0                        // layer-0 spikes [16][HSROW] f16
#define OFF_HS1   (BT*HSROW)               // layer-1 spikes
#define OFF_XT    (2*BT*HSROW)             // x double buffer [2][16] v4f
#define OFF_FEAT  (OFF_XT + 2*BT*16)       // feat [16][192] f32
#define LDS_BYTES (OFF_FEAT + BT*HH*4)     // 25600 total

union Frag { v16h h; v4u u[2]; };

#define WMMA(A, B, C) __builtin_amdgcn_wmma_f32_16x16x32_f16( \
    false, (A), false, (B), (short)0, (C), false, false)

__global__ __launch_bounds__(TPB, 1)
void spikernn_har_kernel(const float* __restrict__ x,
                         const float* __restrict__ W_in,
                         const float* __restrict__ b_in,
                         const float* __restrict__ W_cells,
                         const float* __restrict__ b_cells,
                         const float* __restrict__ W_head,
                         const float* __restrict__ b_head,
                         float* __restrict__ out)
{
    extern __shared__ char smem[];
    const int tid  = threadIdx.x;
    const int lane = tid & 31;
    const int wave = tid >> 5;            // 0..11 == N tile index
    const int batchBase = blockIdx.x * BT;
    const int nloc = lane & 15;           // D layout: N = wave*16 + nloc
    const int half = lane >> 4;           // D layout: M = r + 8*half
    const int n = wave * 16 + nloc;       // global hidden index owned by lane

    // zero both spike buffers (contiguous at offset 0)
    for (int i = tid; i < (2 * BT * HSROW) / 4; i += TPB)
        ((unsigned*)(smem + OFF_HS0))[i] = 0u;

    // per-lane time-invariant scalars, straight from global (one-time reads)
    const float win0 = W_in[n*3+0], win1 = W_in[n*3+1], win2 = W_in[n*3+2];
    const float cini0 = b_in[n] + b_cells[n];     // b_in + b_cells[layer0]
    const float bc1   = b_cells[HH + n];          // b_cells[layer1]

    // Register-resident WMMA B fragments for both layers (96 VGPRs total).
    // B frag lane holds column n; elements e=0..15 are K = kb*32+16*half+e,
    // i.e. 16 CONTIGUOUS floats of W_cells[l][n][k].
    v16h bw0[KB], bw1[KB];
    #pragma unroll
    for (int kb = 0; kb < KB; ++kb) {
        const float* s0 = W_cells + (size_t)n        * HH + kb*32 + half*16;
        const float* s1 = W_cells + (size_t)(HH + n) * HH + kb*32 + half*16;
        v16h b0, b1;
        #pragma unroll
        for (int e = 0; e < 16; ++e) { b0[e] = (_Float16)s0[e]; b1[e] = (_Float16)s1[e]; }
        bw0[kb] = b0; bw1[kb] = b1;
    }

    // x prologue: stage t=0 into buffer 0
    const float* xrow = x + (size_t)(batchBase + tid) * TT * 3;  // used if tid<16
    float xr0 = 0.f, xr1 = 0.f, xr2 = 0.f;
    if (tid < BT) {
        xr0 = xrow[0]; xr1 = xrow[1]; xr2 = xrow[2];
        v4f v; v.x = xr0; v.y = xr1; v.z = xr2; v.w = 0.f;
        ((v4f*)(smem + OFF_XT))[tid] = v;
    }
    __syncthreads();

    // membrane potentials + spike-rate accumulator in VGPRs
    float v0[8], v1[8], sAcc[8], sReg[8];
    #pragma unroll
    for (int r = 0; r < 8; ++r) { v0[r] = 0.f; v1[r] = 0.f; sAcc[r] = 0.f; }

    const char* hs0 = smem + OFF_HS0;
    const char* hs1 = smem + OFF_HS1;
    // A-fragment base byte address for this lane: row = nloc (M tile 0),
    // elements e -> K = kb*32 + 16*half + e
    const char* ab0 = hs0 + nloc * HSROW + half * 32;
    const char* ab1 = hs1 + nloc * HSROW + half * 32;

    for (int t = 0; t < TT; ++t) {
        const int cur = t & 1;
        // stage x(t+1) into regs; latency hidden behind this step's compute
        if (tid < BT && (t + 1) < TT) {
            const float* xp = xrow + (size_t)(t + 1) * 3;
            xr0 = xp[0]; xr1 = xp[1]; xr2 = xp[2];
            if ((t + 2) < TT) __builtin_prefetch(xrow + (size_t)(t + 2) * 3, 0, 0);
        }

        // ===== layer 0: z = x@W_in^T + b_in + h0@W0^T + b0 =====
        // issue ALL spike A-fragment loads up front (independent of VALU below)
        Frag af[KB];
        #pragma unroll
        for (int kb = 0; kb < KB; ++kb) {
            af[kb].u[0] = *(const v4u*)(ab0 + kb * 64);
            af[kb].u[1] = *(const v4u*)(ab0 + kb * 64 + 16);
        }
        // chain B starts from zero -> its WMMAs are independent of the cInit
        // VALU below and co-execute with it
        v8f accB = {};
        accB = WMMA(af[0].h, bw0[0], accB);
        v8f accA;
        {
            const v4f* xtp = (const v4f*)(smem + OFF_XT) + cur * BT + half * 8;
            #pragma unroll
            for (int r = 0; r < 8; ++r) {
                v4f v = xtp[r];
                accA[r] = v.x * win0 + v.y * win1 + v.z * win2 + cini0;
            }
        }
        // two interleaved depth-3 accumulation chains (deps are 2-back)
        accA = WMMA(af[1].h, bw0[1], accA);
        accB = WMMA(af[2].h, bw0[2], accB);
        accA = WMMA(af[3].h, bw0[3], accA);
        accB = WMMA(af[4].h, bw0[4], accB);
        accA = WMMA(af[5].h, bw0[5], accA);
        #pragma unroll
        for (int r = 0; r < 8; ++r) {      // merge + LIF: charge, fire, reset
            float vm = 0.5f * (v0[r] + (accA[r] + accB[r]));
            bool fire = vm >= 1.f;
            v0[r] = fire ? 0.f : vm;
            sReg[r] = fire ? 1.f : 0.f;
        }
        __syncthreads();  // sync1: hs0 + XT[cur] reads done (fences writes below)

        // layer-1 A fragments (h1 from t-1; their last write was fenced by sync1)
        #pragma unroll
        for (int kb = 0; kb < KB; ++kb) {
            af[kb].u[0] = *(const v4u*)(ab1 + kb * 64);
            af[kb].u[1] = *(const v4u*)(ab1 + kb * 64 + 16);
        }
        // publish s0 spikes (next step's layer-0 A); visible after sync2
        #pragma unroll
        for (int r = 0; r < 8; ++r)
            *(_Float16*)(hs0 + (r + 8 * half) * HSROW + n * 2) = (_Float16)sReg[r];
        // publish staged x(t+1); read at top of t+1, fenced by sync2
        if (tid < BT && (t + 1) < TT) {
            v4f v; v.x = xr0; v.y = xr1; v.z = xr2; v.w = 0.f;
            ((v4f*)(smem + OFF_XT))[((t + 1) & 1) * BT + tid] = v;
        }

        // ===== layer 1: z = s0 + h1@W1^T + b1 (s0 is in our own regs) =====
        v8f accB1 = {};
        accB1 = WMMA(af[0].h, bw1[0], accB1);
        v8f accA1;
        #pragma unroll
        for (int r = 0; r < 8; ++r) accA1[r] = sReg[r] + bc1;
        accA1 = WMMA(af[1].h, bw1[1], accA1);
        accB1 = WMMA(af[2].h, bw1[2], accB1);
        accA1 = WMMA(af[3].h, bw1[3], accA1);
        accB1 = WMMA(af[4].h, bw1[4], accB1);
        accA1 = WMMA(af[5].h, bw1[5], accA1);
        #pragma unroll
        for (int r = 0; r < 8; ++r) {
            float vm = 0.5f * (v1[r] + (accA1[r] + accB1[r]));
            bool fire = vm >= 1.f;
            v1[r] = fire ? 0.f : vm;
            float s = fire ? 1.f : 0.f;
            sReg[r] = s;
            sAcc[r] += s;                  // readout accumulation (mean over t)
        }
        __syncthreads();  // sync2: hs1 reads done; hs0/XT writes now visible
        // publish s1 spikes (next step's layer-1 A); reads at t+1 are after sync1
        #pragma unroll
        for (int r = 0; r < 8; ++r)
            *(_Float16*)(hs1 + (r + 8 * half) * HSROW + n * 2) = (_Float16)sReg[r];
    }

    // ---------------- readout mean + head GEMM ----------------
    #pragma unroll
    for (int r = 0; r < 8; ++r)
        ((float*)(smem + OFF_FEAT))[(r + 8 * half) * HH + n] = sAcc[r] * (1.f / (float)TT);
    __syncthreads();
    for (int o = tid; o < BT * 12; o += TPB) {
        int b = o / 12, c = o % 12;
        const float* fr = (const float*)(smem + OFF_FEAT) + b * HH;
        const float* wr = W_head + (size_t)c * HH;      // one-time, L2-resident
        float s = b_head[c];
        #pragma unroll 8
        for (int k = 0; k < HH; ++k) s += fr[k] * wr[k];
        out[(size_t)(batchBase + b) * 12 + c] = s;
    }
}

extern "C" void kernel_launch(void* const* d_in, const int* in_sizes, int n_in,
                              void* d_out, int out_size, void* d_ws, size_t ws_size,
                              hipStream_t stream) {
    const float* x       = (const float*)d_in[0];
    const float* W_in    = (const float*)d_in[1];
    const float* b_in    = (const float*)d_in[2];
    const float* W_cells = (const float*)d_in[3];
    const float* b_cells = (const float*)d_in[4];
    const float* W_head  = (const float*)d_in[5];
    const float* b_head  = (const float*)d_in[6];
    float* out = (float*)d_out;
    (void)in_sizes; (void)n_in; (void)out_size; (void)d_ws; (void)ws_size;

    hipLaunchKernelGGL(spikernn_har_kernel, dim3(8), dim3(TPB), LDS_BYTES, stream,
                       x, W_in, b_in, W_cells, b_cells, W_head, b_head, out);
}
